// MyModel_87522843559879
// MI455X (gfx1250) — matrix operand order repaired and checked by reading, not compiled, not run on hardware
//
#include <hip/hip_runtime.h>
#include <hip/hip_bf16.h>

// ---------------------------------------------------------------------------
// Stacked LSTM (9 layers, U=15) fused into ONE kernel. One wave32 owns 16
// batch rows for all layers x all timesteps (recurrence is the bound; the
// only real parallelism is batch -> 32 independent waves).
// Layers 1-8: ONE v_wmma_f32_16x16x32_f16 per gate:
//   B = [W rows 0..14 ; bias row 15 ; U rows 16..30 ; 0 row 31]  (pre-swizzled)
//   A = [h_{k-1}(t) ; 1.0 ; h_k(t-1) ; *]  -- two ds_load_b128, the 1.0 is
//       baked into column 15 of lds_h so it costs nothing at runtime.
//   C = persistent zero vector (no per-step bias broadcast).
// Sigmoid gates use native v_tanh_f32: sigma(z) = 0.5*tanh(z/2)+0.5 with the
// 0.5 pre-folded into the i/f/o weights+biases at fill time.
// ---------------------------------------------------------------------------

typedef _Float16 v16h __attribute__((ext_vector_type(16)));
typedef _Float16 v8h  __attribute__((ext_vector_type(8)));
typedef float    v8f  __attribute__((ext_vector_type(8)));
typedef float    v4f  __attribute__((ext_vector_type(4)));

#define BB 512
#define TT 1024
#define FF 32
#define UU 15
#define NL 9
#define NG 4   // gates i,f,g,o ; each 15 cols, padded to 16

static __device__ __forceinline__ v16h zero16() {
    v16h z;
#pragma unroll
    for (int i = 0; i < 16; ++i) z[i] = (_Float16)0.0f;
    return z;
}

static __device__ __forceinline__ float fast_rcp(float x) {
#if __has_builtin(__builtin_amdgcn_rcpf)
    return __builtin_amdgcn_rcpf(x);
#else
    return 1.0f / x;
#endif
}

static __device__ __forceinline__ float fast_tanh(float x) {
#if __has_builtin(__builtin_amdgcn_tanhf)
    return __builtin_amdgcn_tanhf(x);        // native v_tanh_f32
#else
    return 1.0f - 2.0f * fast_rcp(__expf(2.0f * x) + 1.0f);
#endif
}

__global__ __launch_bounds__(32) void lstm_stack_kernel(
    const float* __restrict__ x,   // [B,T,F]
    const float* __restrict__ W0,  // [F,60]
    const float* __restrict__ U0,  // [15,60]
    const float* __restrict__ b0,  // [60]
    const float* __restrict__ Wk,  // [8,15,60]
    const float* __restrict__ Uk,  // [8,15,60]
    const float* __restrict__ bk,  // [8,60]
    const float* __restrict__ Wd,  // [15,1]
    const float* __restrict__ bd,  // [1]
    float* __restrict__ out)       // [B,T,1]
{
    // B-operand layout (16x16x32 f16): lane L -> N=L%16, K=(L/16)*8 + (j&7) + (j>>3)*16.
    __shared__ __attribute__((aligned(32))) _Float16 lds_bw0[NG][32][16];        // layer0 W (K=x feats 0..31)
    __shared__ __attribute__((aligned(32))) _Float16 lds_bu0[NG][32][16];        // layer0: U rows 0..14, bias row 15
    __shared__ __attribute__((aligned(32))) _Float16 lds_bs[NL - 1][NG][32][16]; // layers>=1: W 0..14, bias 15, U 16..30
    __shared__ __attribute__((aligned(16))) _Float16 lds_h[NL][16][16];          // h; col 15 == 1.0 (bias lane), fixed
    __shared__ float lds_wd[16];
    __shared__ float lds_bd;

    const int lane = threadIdx.x;       // 0..31
    const int base = blockIdx.x * 16;   // batch tile
    const int m    = lane & 15;         // row (A) / col N (B, C/D)
    const int gr   = lane >> 4;         // lane group

    // ------------- one-time weight convert + swizzle (0.5 pre-folded) -------
    // gate g==2 ('g' gate, tanh) unscaled; i/f/o scaled by 0.5 for the
    // sigmoid-as-tanh identity.
    for (int idx = lane; idx < NG * 32 * 16; idx += 32) {            // layer-0 W and U(+bias)
        int j = idx & 15, l = (idx >> 4) & 31, g = idx >> 9;
        int K = ((l >> 4) << 3) + (j & 7) + ((j >> 3) << 4), n = l & 15;
        float sc = (g == 2) ? 1.0f : 0.5f;
        int col = g * UU + n;
        float vw = (n < UU) ? sc * W0[K * (NG * UU) + col] : 0.0f;
        float vu = 0.0f;
        if (n < UU) {
            if (K < UU)       vu = sc * U0[K * (NG * UU) + col];
            else if (K == UU) vu = sc * b0[col];                     // bias row
        }
        lds_bw0[g][l][j] = (_Float16)vw;
        lds_bu0[g][l][j] = (_Float16)vu;
    }
    for (int idx = lane; idx < (NL - 1) * NG * 32 * 16; idx += 32) { // layers 1..8: [W ; bias ; U ; 0]
        int j = idx & 15, l = (idx >> 4) & 31, g = (idx >> 9) & 3, k8 = idx >> 11;
        int n = l & 15, K = ((l >> 4) << 3) + (j & 7) + ((j >> 3) << 4);
        float sc = (g == 2) ? 1.0f : 0.5f;
        float v = 0.0f;
        if (n < UU) {
            int col = g * UU + n;
            if (K < 16) {
                if (K < UU)       v = sc * Wk[(k8 * UU + K) * (NG * UU) + col];
                else if (K == UU) v = sc * bk[k8 * (NG * UU) + col]; // bias row K=15
            } else {
                if (K - 16 < UU)  v = sc * Uk[(k8 * UU + (K - 16)) * (NG * UU) + col];
            }
        }
        lds_bs[k8][g][l][j] = (_Float16)v;
    }
    if (lane < 16) lds_wd[lane] = (lane < UU) ? Wd[lane] : 0.0f;
    if (lane == 0) lds_bd = bd[0];
    for (int idx = lane; idx < NL * 16 * 16; idx += 32)              // h = 0 ; col 15 = 1.0 (bias source)
        ((_Float16*)lds_h)[idx] = ((idx & 15) == UU) ? (_Float16)1.0f : (_Float16)0.0f;
    __syncthreads();

    // persistent zero accumulator (loop-invariant, D != C encoding -> no copies)
    v8f czero;
#pragma unroll
    for (int i = 0; i < 8; ++i) czero[i] = 0.0f;

    // recurrent c-state in registers (C/D layout: lane holds M=gr*8+r, N=m)
    float c[NL][8];
#pragma unroll
    for (int k = 0; k < NL; ++k)
#pragma unroll
        for (int r = 0; r < 8; ++r) c[k][r] = 0.0f;

    const float* xrow = x + (size_t)(base + m) * TT * FF;

#pragma unroll 1
    for (int t = 0; t < TT; ++t) {
        // ======== layer 0: A from x (K=32), W-WMMA then U(+bias)-WMMA ========
        v16h Ax;
        {
            const float* p0 = xrow + (size_t)t * FF + gr * 8;
            v4f a0 = *(const v4f*)(p0);
            v4f a1 = *(const v4f*)(p0 + 4);
            v4f a2 = *(const v4f*)(p0 + 16);
            v4f a3 = *(const v4f*)(p0 + 20);
#pragma unroll
            for (int i = 0; i < 4; ++i) {
                Ax[i]      = (_Float16)a0[i];
                Ax[4 + i]  = (_Float16)a1[i];
                Ax[8 + i]  = (_Float16)a2[i];
                Ax[12 + i] = (_Float16)a3[i];
            }
        }
        v16h Ah0 = zero16();                 // elements 0..7 carry h0(t-1) + baked 1.0 at K=15
        {
            v8h hl = *(const v8h*)&lds_h[0][m][gr * 8];
#pragma unroll
            for (int i = 0; i < 8; ++i) Ah0[i] = hl[i];
        }
        {
            v8f acc[NG];
#pragma unroll
            for (int g4 = 0; g4 < NG; ++g4) {
                v8f a = __builtin_amdgcn_wmma_f32_16x16x32_f16(
                            false, Ax, false, *(const v16h*)&lds_bw0[g4][lane][0],
                            (short)0, czero, false, false);
                acc[g4] = __builtin_amdgcn_wmma_f32_16x16x32_f16(
                            false, Ah0, false, *(const v16h*)&lds_bu0[g4][lane][0],
                            (short)0, a, false, false);
            }
            _Float16 hh[8];
#pragma unroll
            for (int r = 0; r < 8; ++r) {
                float iv = fmaf(0.5f, fast_tanh(acc[0][r]), 0.5f);
                float fv = fmaf(0.5f, fast_tanh(acc[1][r]), 0.5f);
                float gv = fast_tanh(acc[2][r]);
                float ov = fmaf(0.5f, fast_tanh(acc[3][r]), 0.5f);
                float cn = fmaf(fv, c[0][r], iv * gv);
                c[0][r] = cn;
                hh[r] = (_Float16)(ov * fast_tanh(cn));
            }
            if (m != UU) {                   // never touch the baked-1.0 bias column
#pragma unroll
                for (int r = 0; r < 8; ++r)
                    lds_h[0][(gr << 3) + r][m] = hh[r];
            }
        }

        // ======== layers 1..8: single WMMA per gate, bias fused ========
#pragma unroll
        for (int k = 1; k < NL; ++k) {
            v16h A;
            {
                // elems 0..7  -> K=gr*8+j    : h_{k-1}(t) (+1.0 at K=15)
                // elems 8..15 -> K=16+gr*8+j : h_k(t-1)   (K=31 row of B is 0)
                v8h hlo = *(const v8h*)&lds_h[k - 1][m][gr * 8];
                v8h hhi = *(const v8h*)&lds_h[k][m][gr * 8];
#pragma unroll
                for (int i = 0; i < 8; ++i) { A[i] = hlo[i]; A[8 + i] = hhi[i]; }
            }
            v8f acc[NG];
#pragma unroll
            for (int g4 = 0; g4 < NG; ++g4) {
                acc[g4] = __builtin_amdgcn_wmma_f32_16x16x32_f16(
                              false, A, false, *(const v16h*)&lds_bs[k - 1][g4][lane][0],
                              (short)0, czero, false, false);
            }
            _Float16 hh[8];
#pragma unroll
            for (int r = 0; r < 8; ++r) {
                float iv = fmaf(0.5f, fast_tanh(acc[0][r]), 0.5f);
                float fv = fmaf(0.5f, fast_tanh(acc[1][r]), 0.5f);
                float gv = fast_tanh(acc[2][r]);
                float ov = fmaf(0.5f, fast_tanh(acc[3][r]), 0.5f);
                float cn = fmaf(fv, c[k][r], iv * gv);
                c[k][r] = cn;
                hh[r] = (_Float16)(ov * fast_tanh(cn));
            }
            if (m != UU) {
#pragma unroll
                for (int r = 0; r < 8; ++r)
                    lds_h[k][(gr << 3) + r][m] = hh[r];
            }
        }

        // ======== fused dense head: out[b,t] = h_8(t) . Wd + bd ========
        if (lane < 16) {
            float s = lds_bd;
#pragma unroll
            for (int n = 0; n < UU; ++n)
                s += (float)lds_h[NL - 1][lane][n] * lds_wd[n];
            out[(size_t)(base + lane) * TT + t] = s;
        }
    }
}

extern "C" void kernel_launch(void* const* d_in, const int* in_sizes, int n_in,
                              void* d_out, int out_size, void* d_ws, size_t ws_size,
                              hipStream_t stream) {
    const float* x  = (const float*)d_in[0];
    const float* W0 = (const float*)d_in[1];
    const float* U0 = (const float*)d_in[2];
    const float* b0 = (const float*)d_in[3];
    const float* Wk = (const float*)d_in[4];
    const float* Uk = (const float*)d_in[5];
    const float* bk = (const float*)d_in[6];
    const float* Wd = (const float*)d_in[7];
    const float* bd = (const float*)d_in[8];
    float* out = (float*)d_out;

    lstm_stack_kernel<<<BB / 16, 32, 0, stream>>>(x, W0, U0, b0, Wk, Uk, bk, Wd, bd, out);
}
